// MultiHeadAttention_33887291965464
// MI455X (gfx1250) — matrix operand order repaired and checked by reading, not compiled, not run on hardware
//
#include <hip/hip_runtime.h>
#include <hip/hip_bf16.h>

// ProbSparse attention (Informer) for B=2, L=2048, D=512, H=8, d=64.
// U_part = 2048 sampled keys, u = 40 selected queries (padded to 48 = 3 WMMA tiles).
// GEMMs via v_wmma_f32_16x16x32_bf16; tile staging via global_load_async_to_lds_b128
// (ASYNCcnt) + s_wait_asynccnt; look-ahead via global_prefetch.

#define BB 2
#define LL 2048
#define DD 512
#define HH 8
#define DH 64
#define UPART 2048
#define UTOP 40
#define UPAD 48

typedef __attribute__((ext_vector_type(16))) __bf16 v16bf;
typedef __attribute__((ext_vector_type(8)))  float  v8f;

union FragAB { v16bf v; unsigned int u[8]; };
union FragC  { v8f  v; float f[8]; };

__device__ __forceinline__ unsigned short f32_to_bf16(float f) {
  unsigned int x = __builtin_bit_cast(unsigned int, f);
  x += 0x7FFFu + ((x >> 16) & 1u);          // round-to-nearest-even
  return (unsigned short)(x >> 16);
}

// ---- CDNA5 async memory->LDS copy (ASYNCcnt path, ISA 15.18.3 op 98) -------
__device__ __forceinline__ unsigned lds_addr_of(const void* p) {
  return (unsigned)(size_t)(__attribute__((address_space(3))) const char*)p;
}
__device__ __forceinline__ void async_copy_b128(const unsigned short* gsrc,
                                                unsigned short* ldst) {
  unsigned lds = lds_addr_of(ldst);
  unsigned long long ga = (unsigned long long)(size_t)gsrc;
  asm volatile("global_load_async_to_lds_b128 %0, %1, off"
               :: "v"(lds), "v"(ga) : "memory");
}
__device__ __forceinline__ void wait_async0() {
  asm volatile("s_wait_asynccnt 0x0" ::: "memory");
}

// A fragment: 16x32 bf16, source row-major [M][ld] (K contiguous).
__device__ __forceinline__ v16bf load_frag_a(const unsigned short* base, int ld, int lane) {
  int m = lane & 15, half = lane >> 4;
  const unsigned short* p = base + (size_t)m * ld;
  FragAB f;
#pragma unroll
  for (int j = 0; j < 8; ++j) {
    int kk = ((j < 4) ? 0 : 16) + half * 8 + (j & 3) * 2;
    f.u[j] = *(const unsigned int*)(p + kk);
  }
  return f.v;
}

// B fragment: 32x16 bf16, B[k][n]; source [N][ld] (K contiguous per N row).
__device__ __forceinline__ v16bf load_frag_b(const unsigned short* base, int ld, int lane) {
  int n = lane & 15, half = lane >> 4;
  const unsigned short* p = base + (size_t)n * ld;
  FragAB f;
#pragma unroll
  for (int j = 0; j < 8; ++j) {
    int kk = ((j < 4) ? 0 : 16) + half * 8 + (j & 3) * 2;
    f.u[j] = *(const unsigned int*)(p + kk);
  }
  return f.v;
}

__device__ __forceinline__ v8f wmma_bf16(v16bf a, v16bf b, v8f c) {
  return __builtin_amdgcn_wmma_f32_16x16x32_bf16(false, a, false, b, (short)0, c, false, false);
}

// ---------------------------------------------------------------- kernels ---

__global__ void cvt_f32_bf16(const float* __restrict__ src,
                             unsigned short* __restrict__ dst, int n) {
  for (int i = blockIdx.x * blockDim.x + threadIdx.x; i < n; i += gridDim.x * blockDim.x)
    dst[i] = f32_to_bf16(src[i]);
}

// Y(4096,512) = X(4096,512) @ W^T(512,512) + bias, bf16 out.
// Block 256 thr = 8 waves; 64x64 tile; waves 4(M) x 2(N), 16x32 each.
// Both 64x32 K-step tiles staged by async copies (1 b128 chunk per thread per tile).
__global__ void proj_gemm(const unsigned short* __restrict__ X,
                          const unsigned short* __restrict__ W,
                          const float* __restrict__ bias,
                          unsigned short* __restrict__ Y) {
  __shared__ unsigned short Xs[64 * 32];
  __shared__ unsigned short Ws[64 * 32];
  const int tid = threadIdx.x, lane = tid & 31, wave = tid >> 5;
  const int m0 = blockIdx.x * 64, n0 = blockIdx.y * 64;
  const int wm = wave >> 1, wn = wave & 1;
  const int srow = tid >> 2, soff = (tid & 3) * 8;   // 256 x 16B chunks per tile
  v8f acc0 = {}, acc1 = {};
  for (int kb = 0; kb < DD; kb += 32) {
    __syncthreads();                                 // previous tile consumed
    async_copy_b128(X + (size_t)(m0 + srow) * DD + kb + soff, Xs + srow * 32 + soff);
    async_copy_b128(W + (size_t)(n0 + srow) * DD + kb + soff, Ws + srow * 32 + soff);
    if (kb + 32 < DD) {                              // prefetch next K step into GL2
      __builtin_prefetch(X + (size_t)(m0 + srow) * DD + kb + 32 + soff, 0, 0);
      __builtin_prefetch(W + (size_t)(n0 + srow) * DD + kb + 32 + soff, 0, 0);
    }
    wait_async0();
    __syncthreads();
    v16bf a  = load_frag_a(Xs + (wm * 16) * 32, 32, lane);
    v16bf b0 = load_frag_b(Ws + (wn * 32) * 32, 32, lane);
    v16bf b1 = load_frag_b(Ws + (wn * 32 + 16) * 32, 32, lane);
    acc0 = wmma_bf16(a, b0, acc0);
    acc1 = wmma_bf16(a, b1, acc1);
  }
  int half = lane >> 4, n = lane & 15;
  FragC c0, c1; c0.v = acc0; c1.v = acc1;
#pragma unroll
  for (int r = 0; r < 8; ++r) {
    int row = m0 + wm * 16 + half * 8 + r;
    int col0 = n0 + wn * 32 + n, col1 = col0 + 16;
    Y[(size_t)row * DD + col0] = f32_to_bf16(c0.f[r] + bias[col0]);
    Y[(size_t)row * DD + col1] = f32_to_bf16(c1.f[r] + bias[col1]);
  }
}

// Ks[bh][s][dd] = kp[b, index_sample[s], h*64+dd]  (16B per thread)
__global__ void gather_ks(const unsigned short* __restrict__ kp,
                          const int* __restrict__ idx,
                          unsigned short* __restrict__ Ks) {
  int t = blockIdx.x * blockDim.x + threadIdx.x;
  if (t >= BB * HH * UPART * 8) return;
  int chunk = t & 7;
  int s = (t >> 3) & (UPART - 1);
  int bh = t >> 14;
  int b = bh >> 3, h = bh & 7;
  int row = idx[s];
  *(uint4*)(Ks + ((size_t)bh * UPART + s) * DH + chunk * 8) =
      *(const uint4*)(kp + ((size_t)b * LL + row) * DD + h * DH + chunk * 8);
}

// M[bh][q] = max_s(q.Ks) - sum_s(q.Ks)/L_K ; streaming over s, never storing QKs.
// Block 256 = 8 waves, one 16-row q tile each; Ks chunks (128x64) async-staged in LDS.
__global__ void qks_m_kernel(const unsigned short* __restrict__ qp,
                             const unsigned short* __restrict__ Ks,
                             float* __restrict__ M) {
  __shared__ unsigned short Sk[128 * DH];            // 16 KB
  const int tid = threadIdx.x, lane = tid & 31, wave = tid >> 5;
  const int bh = blockIdx.y, b = bh >> 3, h = bh & 7;
  const int q0 = blockIdx.x * 128 + wave * 16;
  // qh[b,h,q,:] is the flat reshape of qp: offset b*L*D + h*L*dh + q*dh
  const unsigned short* qbase = qp + (size_t)b * (LL * DD) + (size_t)h * (LL * DH) + (size_t)q0 * DH;
  v16bf a0 = load_frag_a(qbase, DH, lane);
  v16bf a1 = load_frag_a(qbase + 32, DH, lane);
  float lmax[8], lsum[8];
#pragma unroll
  for (int r = 0; r < 8; ++r) { lmax[r] = -3.4e38f; lsum[r] = 0.f; }
  const unsigned short* kbase = Ks + (size_t)bh * (UPART * DH);
  for (int c = 0; c < UPART / 128; ++c) {
    __syncthreads();
#pragma unroll
    for (int q = 0; q < 4; ++q) {                    // 1024 x 16B = 128x64 bf16
      int id = tid * 4 + q;
      int row = id >> 3, off = (id & 7) * 8;
      async_copy_b128(kbase + (size_t)(c * 128 + row) * DH + off, Sk + row * DH + off);
    }
    if (c + 1 < UPART / 128)                         // prefetch next chunk into GL2
      __builtin_prefetch(kbase + (size_t)((c + 1) * 128 + (tid >> 1)) * DH + (tid & 1) * 32, 0, 0);
    wait_async0();
    __syncthreads();
#pragma unroll
    for (int st = 0; st < 8; ++st) {
      v16bf b0 = load_frag_b(Sk + st * 16 * DH, DH, lane);
      v16bf b1 = load_frag_b(Sk + st * 16 * DH + 32, DH, lane);
      v8f acc = {};
      acc = wmma_bf16(a0, b0, acc);
      acc = wmma_bf16(a1, b1, acc);
      FragC cc; cc.v = acc;
#pragma unroll
      for (int r = 0; r < 8; ++r) {
        lmax[r] = fmaxf(lmax[r], cc.f[r]);
        lsum[r] += cc.f[r];
      }
    }
  }
#pragma unroll
  for (int r = 0; r < 8; ++r) {                      // reduce over 16 N-lanes
#pragma unroll
    for (int msk = 1; msk < 16; msk <<= 1) {
      lmax[r] = fmaxf(lmax[r], __shfl_xor(lmax[r], msk, 32));
      lsum[r] += __shfl_xor(lsum[r], msk, 32);
    }
  }
  if ((lane & 15) == 0) {
    int half = lane >> 4;
#pragma unroll
    for (int r = 0; r < 8; ++r)
      M[(size_t)bh * LL + q0 + half * 8 + r] = lmax[r] - lsum[r] * (1.0f / (float)UPART);
  }
}

// 40 smallest M per (b,h) (== top_k of -M), ascending, ties -> lower index.
__global__ void topk_kernel(const float* __restrict__ M, int* __restrict__ Mtop) {
  __shared__ float vals[LL];
  __shared__ float rv[256];
  __shared__ int   ri[256];
  const int tid = threadIdx.x, bh = blockIdx.x;
  const float* Mrow = M + (size_t)bh * LL;
  for (int i = tid; i < LL; i += 256) vals[i] = Mrow[i];
  __syncthreads();
  for (int t = 0; t < UTOP; ++t) {
    float bv = 3.4e38f; int bi = 0x7FFFFFFF;
    for (int i = tid; i < LL; i += 256) {
      float v = vals[i];
      if (v < bv) { bv = v; bi = i; }
    }
    rv[tid] = bv; ri[tid] = bi;
    __syncthreads();
    for (int s = 128; s > 0; s >>= 1) {
      if (tid < s) {
        float ov = rv[tid + s]; int oi = ri[tid + s];
        if (ov < rv[tid] || (ov == rv[tid] && oi < ri[tid])) { rv[tid] = ov; ri[tid] = oi; }
      }
      __syncthreads();
    }
    if (tid == 0) { Mtop[bh * UPAD + t] = ri[0]; vals[ri[0]] = 3.4e38f; }
    __syncthreads();
  }
  if (tid < UPAD - UTOP) Mtop[bh * UPAD + UTOP + tid] = 0;
}

// Qred[bh][i][:] = qh row Mtop[bh][i]; pad rows 40..47 with zeros.
__global__ void gather_qred(const unsigned short* __restrict__ qp,
                            const int* __restrict__ Mtop,
                            unsigned short* __restrict__ Qred) {
  int t = blockIdx.x * blockDim.x + threadIdx.x;
  if (t >= BB * HH * UPAD * 8) return;
  int chunk = t & 7;
  int i = (t >> 3) % UPAD;
  int bh = t / (UPAD * 8);
  int b = bh >> 3, h = bh & 7;
  uint4* dst = (uint4*)(Qred + ((size_t)bh * UPAD + i) * DH + chunk * 8);
  if (i < UTOP) {
    int q = Mtop[bh * UPAD + i];
    *dst = *(const uint4*)(qp + (size_t)b * (LL * DD) + (size_t)h * (LL * DH)
                              + (size_t)q * DH + chunk * 8);
  } else {
    uint4 z = {0u, 0u, 0u, 0u};
    *dst = z;
  }
}

// scores[bh][i][s] = (Qred . kh) / 8 ; block 96 thr = 3 waves (one q-tile each).
__global__ void scores_kernel(const unsigned short* __restrict__ Qred,
                              const unsigned short* __restrict__ kp,
                              float* __restrict__ scores) {
  const int tid = threadIdx.x, lane = tid & 31, wave = tid >> 5;   // 0..2
  const int bh = blockIdx.x, b = bh >> 3, h = bh & 7;
  const unsigned short* qb = Qred + ((size_t)bh * UPAD + wave * 16) * DH;
  v16bf a0 = load_frag_a(qb, DH, lane);
  v16bf a1 = load_frag_a(qb + 32, DH, lane);
  const unsigned short* kb = kp + (size_t)b * (LL * DD) + h * DH;  // kh[b,h,s,k] = kp[b,s,h*64+k]
  float* srow = scores + (size_t)(bh * UPAD + wave * 16) * LL;
  const int half = lane >> 4, n = lane & 15;
  for (int st = 0; st < LL / 16; ++st) {
    const unsigned short* kbs = kb + (size_t)(st * 16) * DD;
    if (st + 4 < LL / 16)
      __builtin_prefetch(kb + (size_t)((st + 4) * 16 + (lane & 15)) * DD + (lane >> 4) * 32, 0, 0);
    v16bf b0 = load_frag_b(kbs, DD, lane);
    v16bf b1 = load_frag_b(kbs + 32, DD, lane);
    v8f acc = {};
    acc = wmma_bf16(a0, b0, acc);
    acc = wmma_bf16(a1, b1, acc);
    FragC cc; cc.v = acc;
#pragma unroll
    for (int r = 0; r < 8; ++r)
      srow[(size_t)(half * 8 + r) * LL + st * 16 + n] = cc.f[r] * 0.125f;
  }
}

// Row softmax over 2048 cols; probs written as bf16 for the PV WMMA.
__global__ void softmax_kernel(const float* __restrict__ scores,
                               unsigned short* __restrict__ probs) {
  __shared__ float red[256];
  const int tid = threadIdx.x;
  const float* s = scores + (size_t)blockIdx.x * LL;
  float m = -3.4e38f;
  for (int i = tid; i < LL; i += 256) m = fmaxf(m, s[i]);
  red[tid] = m; __syncthreads();
  for (int q = 128; q > 0; q >>= 1) { if (tid < q) red[tid] = fmaxf(red[tid], red[tid + q]); __syncthreads(); }
  m = red[0]; __syncthreads();
  float sum = 0.f;
  for (int i = tid; i < LL; i += 256) sum += __expf(s[i] - m);
  red[tid] = sum; __syncthreads();
  for (int q = 128; q > 0; q >>= 1) { if (tid < q) red[tid] += red[tid + q]; __syncthreads(); }
  float inv = 1.0f / red[0];
  unsigned short* p = probs + (size_t)blockIdx.x * LL;
  for (int i = tid; i < LL; i += 256) p[i] = f32_to_bf16(__expf(s[i] - m) * inv);
}

// attn[b][i][h*64+dd] = sum_s probs[bh][i][s] * vh[b,h,s,dd].
// Block 384 thr = 12 waves = 3 i-tiles x 4 n-tiles; V transposed through LDS.
__global__ void pv_kernel(const unsigned short* __restrict__ probs,
                          const unsigned short* __restrict__ vp,
                          unsigned short* __restrict__ attn) {
  __shared__ unsigned short Vt[DH * 32];             // [dd][s_local]
  const int tid = threadIdx.x, lane = tid & 31, wave = tid >> 5;   // 0..11
  const int bh = blockIdx.x, b = bh >> 3, h = bh & 7;
  const int it = wave >> 2, nt = wave & 3;
  const unsigned short* pb = probs + (size_t)(bh * UPAD + it * 16) * LL;
  const unsigned short* vb = vp + (size_t)b * (LL * DD) + h * DH;  // vh[b,h,s,dd] = vp[b,s,h*64+dd]
  v8f acc = {};
  for (int c = 0; c < LL / 32; ++c) {
    __syncthreads();
    for (int i = tid; i < DH * 32; i += 384) {       // transpose 32 V rows into LDS
      int sl = i & 31, dd = i >> 5;
      Vt[dd * 32 + sl] = vb[(size_t)(c * 32 + sl) * DD + dd];
    }
    if (c + 1 < LL / 32 && tid < 64)                 // prefetch next 32 V rows
      __builtin_prefetch(vb + (size_t)((c + 1) * 32 + (tid >> 1)) * DD + (tid & 1) * 32, 0, 0);
    __syncthreads();
    v16bf a  = load_frag_a(pb + c * 32, LL, lane);
    v16bf bf = load_frag_b(Vt + nt * 16 * 32, 32, lane);
    acc = wmma_bf16(a, bf, acc);
  }
  const int half = lane >> 4, n = lane & 15;
  FragC cc; cc.v = acc;
#pragma unroll
  for (int r = 0; r < 8; ++r) {
    int i = it * 16 + half * 8 + r;
    attn[((size_t)b * UPAD + i) * DD + h * DH + nt * 16 + n] = f32_to_bf16(cc.f[r]);
  }
}

// out(B,40,512) = attn(B,48,512) @ Wo^T + bo ; drop pad rows on store.
__global__ void out_proj(const unsigned short* __restrict__ attn,
                         const unsigned short* __restrict__ Wo,
                         const float* __restrict__ bo,
                         float* __restrict__ out) {
  const int tid = threadIdx.x, lane = tid & 31, wave = tid >> 5;
  const int w = blockIdx.x * 8 + wave;               // 0..191 = 6 m-tiles x 32 n-tiles
  const int mtile = w >> 5, ntile = w & 31;
  const unsigned short* ab = attn + (size_t)(mtile * 16) * DD;
  const unsigned short* wb = Wo + (size_t)(ntile * 16) * DD;
  v8f acc = {};
  for (int kb = 0; kb < DD; kb += 32) {
    v16bf a  = load_frag_a(ab + kb, DD, lane);
    v16bf bb = load_frag_b(wb + kb, DD, lane);
    acc = wmma_bf16(a, bb, acc);
  }
  const int half = lane >> 4, n = lane & 15;
  FragC cc; cc.v = acc;
  const int col = ntile * 16 + n;
  const float bias = bo[col];
#pragma unroll
  for (int r = 0; r < 8; ++r) {
    int m = mtile * 16 + half * 8 + r;
    int b = m / UPAD, i = m % UPAD;
    if (i < UTOP) out[((size_t)b * UTOP + i) * DD + col] = cc.f[r] + bias;
  }
}

// ---------------------------------------------------------------- launch ----

extern "C" void kernel_launch(void* const* d_in, const int* in_sizes, int n_in,
                              void* d_out, int out_size, void* d_ws, size_t ws_size,
                              hipStream_t stream) {
  const float* q  = (const float*)d_in[0];
  const float* k  = (const float*)d_in[1];
  const float* v  = (const float*)d_in[2];
  const float* Wq = (const float*)d_in[3];
  const float* bq = (const float*)d_in[4];
  const float* Wk = (const float*)d_in[5];
  const float* bk = (const float*)d_in[6];
  const float* Wv = (const float*)d_in[7];
  const float* bv = (const float*)d_in[8];
  const float* Wo = (const float*)d_in[9];
  const float* bo = (const float*)d_in[10];
  const int* idx  = (const int*)d_in[11];
  float* out = (float*)d_out;

  char* base = (char*)d_ws;
  size_t off = 0;
  auto carve = [&](size_t bytes) -> char* {
    char* p = base + off;
    off = (off + bytes + 255) & ~(size_t)255;
    return p;
  };
  const size_t nX = (size_t)BB * LL * DD;            // 2,097,152
  const size_t nW = (size_t)DD * DD;                 // 262,144
  unsigned short* q_bf  = (unsigned short*)carve(nX * 2);
  unsigned short* k_bf  = (unsigned short*)carve(nX * 2);
  unsigned short* v_bf  = (unsigned short*)carve(nX * 2);
  unsigned short* Wq_bf = (unsigned short*)carve(nW * 2);
  unsigned short* Wk_bf = (unsigned short*)carve(nW * 2);
  unsigned short* Wv_bf = (unsigned short*)carve(nW * 2);
  unsigned short* Wo_bf = (unsigned short*)carve(nW * 2);
  unsigned short* qp_bf = (unsigned short*)carve(nX * 2);
  unsigned short* kp_bf = (unsigned short*)carve(nX * 2);
  unsigned short* vp_bf = (unsigned short*)carve(nX * 2);
  unsigned short* Ks_bf = (unsigned short*)carve((size_t)BB * HH * UPART * DH * 2);
  float*          Mbuf  = (float*)carve((size_t)BB * HH * LL * 4);
  int*            Mtop  = (int*)carve((size_t)BB * HH * UPAD * 4);
  unsigned short* Qred  = (unsigned short*)carve((size_t)BB * HH * UPAD * DH * 2);
  float*          sc    = (float*)carve((size_t)BB * HH * UPAD * LL * 4);
  unsigned short* pr    = (unsigned short*)carve((size_t)BB * HH * UPAD * LL * 2);
  unsigned short* attn  = (unsigned short*)carve((size_t)BB * UPAD * DD * 2);
  (void)ws_size; (void)n_in; (void)in_sizes; (void)out_size;

  // 1) f32 -> bf16 conversions
  cvt_f32_bf16<<<512, 256, 0, stream>>>(q,  q_bf,  (int)nX);
  cvt_f32_bf16<<<512, 256, 0, stream>>>(k,  k_bf,  (int)nX);
  cvt_f32_bf16<<<512, 256, 0, stream>>>(v,  v_bf,  (int)nX);
  cvt_f32_bf16<<<256, 256, 0, stream>>>(Wq, Wq_bf, (int)nW);
  cvt_f32_bf16<<<256, 256, 0, stream>>>(Wk, Wk_bf, (int)nW);
  cvt_f32_bf16<<<256, 256, 0, stream>>>(Wv, Wv_bf, (int)nW);
  cvt_f32_bf16<<<256, 256, 0, stream>>>(Wo, Wo_bf, (int)nW);

  // 2) QKV projections (WMMA, async-staged tiles)
  dim3 gp(BB * LL / 64, DD / 64);
  proj_gemm<<<gp, 256, 0, stream>>>(q_bf, Wq_bf, bq, qp_bf);
  proj_gemm<<<gp, 256, 0, stream>>>(k_bf, Wk_bf, bk, kp_bf);
  proj_gemm<<<gp, 256, 0, stream>>>(v_bf, Wv_bf, bv, vp_bf);

  // 3) gather sampled keys
  gather_ks<<<(BB * HH * UPART * 8 + 255) / 256, 256, 0, stream>>>(kp_bf, idx, Ks_bf);

  // 4) streaming sampled-QK^T with fused max/mean reduction (WMMA, async staging)
  qks_m_kernel<<<dim3(LL / 128, BB * HH), 256, 0, stream>>>(qp_bf, Ks_bf, Mbuf);

  // 5) top-k(40) per (b,h)
  topk_kernel<<<BB * HH, 256, 0, stream>>>(Mbuf, Mtop);

  // 6) gather reduced queries (pad to 48 rows)
  gather_qred<<<(BB * HH * UPAD * 8 + 255) / 256, 256, 0, stream>>>(qp_bf, Mtop, Qred);

  // 7) scores = Qred . K^T / sqrt(64) (WMMA)
  scores_kernel<<<BB * HH, 96, 0, stream>>>(Qred, kp_bf, sc);

  // 8) softmax rows -> bf16 probs
  softmax_kernel<<<BB * HH * UPAD, 256, 0, stream>>>(sc, pr);

  // 9) attn = probs . V (WMMA, V transposed through LDS)
  pv_kernel<<<BB * HH, 384, 0, stream>>>(pr, vp_bf, attn);

  // 10) output projection (WMMA) + bias, drop pad rows
  out_proj<<<24, 256, 0, stream>>>(attn, Wo_bf, bo, out);
}
